// SceneEncoder_68959994904866
// MI455X (gfx1250) — matrix operand order repaired and checked by reading, not compiled, not run on hardware
//
#include <hip/hip_runtime.h>
#include <hip/hip_bf16.h>
#include <math.h>

typedef __attribute__((ext_vector_type(16))) _Float16 v16h;
typedef __attribute__((ext_vector_type(8)))  float    v8f;
typedef _Float16 half_t;
typedef unsigned char u8;

#define BB   2
#define THIST 11
#define NAG  128
#define NMAP 768
#define NPT  20
#define NLGT 32
#define NT   928            // tokens per batch = 768+128+32
#define TT   1856           // total tokens (B*NT), divisible by 16
#define DM   256
#define KNN_ 128

// ---------------- input index map (setup_inputs dict order, params flattened) -
enum {
  IN_MAP_PRE_W = 0, IN_MAP_PRE_B, IN_MAP_MID_W, IN_MAP_MID_B,
  IN_MAP_O1_W, IN_MAP_O1_B, IN_MAP_O2_W, IN_MAP_O2_B,
  IN_AG_W0, IN_AG_B0, IN_AG_W1, IN_AG_B1, IN_AG_W2, IN_AG_B2,
  IN_LT_W0, IN_LT_B0, IN_LT_W1, IN_LT_B1, IN_LT_W2, IN_LT_B2,
  IN_AGENT_PE,                 // 20
  IN_LAYER0 = 21,              // 16 leaves/layer: wq,bq,wk,bk,wv,bv,wo,bo,ln1g,ln1b,w1,b1,w2,b2,ln2g,ln2b
  IN_OUT_W = 21 + 48, IN_OUT_B,   // 69,70
  IN_AGENT_FEAT = 71, IN_AGENT_POS, IN_AGENT_HEAD, IN_AGENT_VALID, IN_AGENT_ID,
  IN_MAP_FEAT, IN_MAP_FVALID, IN_MAP_POS, IN_MAP_HEAD, IN_MAP_VALID,
  IN_TL_FEAT, IN_TL_VALID, IN_TL_POS, IN_TL_HEAD
};

// =============================================================================
// Weight packer: f32 row-major (K x N) -> WMMA-B fragment layout
// packed[kt][nt][lane][e] : lane = (n%16) + 16*((k%32)/16), e = (k%32)%16
// =============================================================================
__global__ void pack_w_kernel(const float* __restrict__ src, half_t* __restrict__ dst,
                              int Kd, int Nd) {
  int i = blockIdx.x * blockDim.x + threadIdx.x;
  if (i >= Kd * Nd) return;
  int k = i / Nd, n = i - k * Nd;
  int kt = k >> 5, kr = k & 31, nt = n >> 4, nr = n & 15;
  int lane = nr + ((kr >> 4) << 4);
  int e = kr & 15;
  dst[((((size_t)kt * (Nd >> 4) + nt) * 32 + lane) << 4) + e] = (half_t)src[i];
}

// =============================================================================
// WMMA GEMM: C[TT x Nd] = A[TT x Kd] (f32) @ Wpacked (f16) + bias, fused
// epilogue. One wave per 16x64 output strip: one A fragment feeds 4 WMMAs per
// K-step (4x A reuse), plus global_prefetch of the next A K-chunk.
// flags: 1=ReLU, 2=+residual, 4=*rowmask
// =============================================================================
__global__ __launch_bounds__(32)
void gemm_wmma_kernel(const float* __restrict__ A, const half_t* __restrict__ Wp,
                      const float* __restrict__ bias, const float* __restrict__ res,
                      const u8* __restrict__ rmask, float* __restrict__ C,
                      int Nd, int Kd, int flags) {
  const int tm  = blockIdx.x * 16;
  const int tn4 = blockIdx.y * 64;           // 4 consecutive 16-col tiles
  const int lane = (int)threadIdx.x;
  const int ml = lane & 15;                  // A row within tile / C column within tile
  const int kh = lane >> 4;                  // K-half selector
  const float*  arow = A + (size_t)(tm + ml) * Kd + kh * 16;
  const int     ntiles = Nd >> 4;
  const half_t* bp0 = Wp + (((size_t)(blockIdx.y * 4) * 32 + lane) << 4);
  const size_t  bstep = (size_t)ntiles * 512;   // halfs per K-tile of packed B
  v8f acc0 = {0.f,0.f,0.f,0.f,0.f,0.f,0.f,0.f};
  v8f acc1 = acc0, acc2 = acc0, acc3 = acc0;
  for (int kt = 0; kt < Kd; kt += 32) {
    const float4 f0 = *(const float4*)(arow + kt);
    const float4 f1 = *(const float4*)(arow + kt + 4);
    const float4 f2 = *(const float4*)(arow + kt + 8);
    const float4 f3 = *(const float4*)(arow + kt + 12);
    // speculative prefetch of next K-chunk of A (global_prefetch_b8)
    __builtin_prefetch(arow + kt + 32, 0, 3);
    v16h a;
    a[0]=(half_t)f0.x;  a[1]=(half_t)f0.y;  a[2]=(half_t)f0.z;  a[3]=(half_t)f0.w;
    a[4]=(half_t)f1.x;  a[5]=(half_t)f1.y;  a[6]=(half_t)f1.z;  a[7]=(half_t)f1.w;
    a[8]=(half_t)f2.x;  a[9]=(half_t)f2.y;  a[10]=(half_t)f2.z; a[11]=(half_t)f2.w;
    a[12]=(half_t)f3.x; a[13]=(half_t)f3.y; a[14]=(half_t)f3.z; a[15]=(half_t)f3.w;
    const half_t* bp = bp0 + (size_t)(kt >> 5) * bstep;
    const v16h b0 = *(const v16h*)(bp);
    const v16h b1 = *(const v16h*)(bp + 512);
    const v16h b2 = *(const v16h*)(bp + 1024);
    const v16h b3 = *(const v16h*)(bp + 1536);
    acc0 = __builtin_amdgcn_wmma_f32_16x16x32_f16(false, a, false, b0, (short)0, acc0, false, false);
    acc1 = __builtin_amdgcn_wmma_f32_16x16x32_f16(false, a, false, b1, (short)0, acc1, false, false);
    acc2 = __builtin_amdgcn_wmma_f32_16x16x32_f16(false, a, false, b2, (short)0, acc2, false, false);
    acc3 = __builtin_amdgcn_wmma_f32_16x16x32_f16(false, a, false, b3, (short)0, acc3, false, false);
  }
  v8f accs[4] = {acc0, acc1, acc2, acc3};
#pragma unroll
  for (int nt = 0; nt < 4; ++nt) {
    const int col = tn4 + nt * 16 + ml;
    const float bv = bias[col];
#pragma unroll
    for (int r = 0; r < 8; ++r) {
      const int row = tm + kh * 8 + r;         // C layout: lanes16-31 -> M+8
      float vv = accs[nt][r] + bv;
      if (flags & 2) vv += res[(size_t)row * Nd + col];
      if (flags & 1) vv = fmaxf(vv, 0.f);
      if (flags & 4) vv = rmask[row] ? vv : 0.f;
      C[(size_t)row * Nd + col] = vv;
    }
  }
}

// =============================================================================
// Agent token MLP: (B,N,352) -> 256 -> 256 -> 256 (+agent_pe), one block/token
// =============================================================================
__global__ __launch_bounds__(256)
void agent_embed_kernel(const float* __restrict__ feat, const u8* __restrict__ mask,
                        const int* __restrict__ aid,
                        const float* w0, const float* b0, const float* w1, const float* b1,
                        const float* w2, const float* b2, const float* pe_tab,
                        float* __restrict__ x) {
  int b = blockIdx.x / NAG, n = blockIdx.x % NAG;
  __shared__ float af[352], h1[256], h2[256];
  int t = threadIdx.x;
  for (int i = t; i < 352; i += 256) {
    int ti = i >> 5, c = i & 31;
    float mv = mask[(b * THIST + ti) * NAG + n] ? 1.f : 0.f;
    af[i] = feat[(((size_t)b * THIST + ti) * NAG + n) * 32 + c] * mv;
  }
  __syncthreads();
  float acc = b0[t];
  for (int k = 0; k < 352; ++k) acc += af[k] * w0[k * DM + t];
  h1[t] = fmaxf(acc, 0.f);
  __syncthreads();
  acc = b1[t];
  for (int k = 0; k < DM; ++k) acc += h1[k] * w1[k * DM + t];
  h2[t] = fmaxf(acc, 0.f);
  __syncthreads();
  acc = b2[t];
  for (int k = 0; k < DM; ++k) acc += h2[k] * w2[k * DM + t];
  int id = aid[b * NAG + n];
  if (id < 0 || id >= NAG) id = NAG - 1;
  acc += pe_tab[id * DM + t];
  x[((size_t)(b * NT + NMAP + n)) * DM + t] = acc;
}

// =============================================================================
// Light token MLP: (B,L,88) -> 256 -> 256 -> 256, one block/token
// =============================================================================
__global__ __launch_bounds__(256)
void light_embed_kernel(const float* __restrict__ feat, const u8* __restrict__ mask,
                        const float* w0, const float* b0, const float* w1, const float* b1,
                        const float* w2, const float* b2, float* __restrict__ x) {
  int b = blockIdx.x / NLGT, n = blockIdx.x % NLGT;
  __shared__ float lf[88], h1[256], h2[256];
  int t = threadIdx.x;
  for (int i = t; i < 88; i += 256) {
    int ti = i >> 3, c = i & 7;
    float mv = mask[(b * THIST + ti) * NLGT + n] ? 1.f : 0.f;
    lf[i] = feat[(((size_t)b * THIST + ti) * NLGT + n) * 8 + c] * mv;
  }
  __syncthreads();
  float acc = b0[t];
  for (int k = 0; k < 88; ++k) acc += lf[k] * w0[k * DM + t];
  h1[t] = fmaxf(acc, 0.f);
  __syncthreads();
  acc = b1[t];
  for (int k = 0; k < DM; ++k) acc += h1[k] * w1[k * DM + t];
  h2[t] = fmaxf(acc, 0.f);
  __syncthreads();
  acc = b2[t];
  for (int k = 0; k < DM; ++k) acc += h2[k] * w2[k * DM + t];
  x[((size_t)(b * NT + NMAP + NAG + n)) * DM + t] = acc;
}

// =============================================================================
// Map PointNet: one 64-thread block per (b,m) polyline of 20 points
// =============================================================================
__global__ __launch_bounds__(64)
void map_embed_kernel(const float* __restrict__ feat, const u8* __restrict__ fmask,
                      const float* prew, const float* preb, const float* midw, const float* midb,
                      const float* o1w, const float* o1b, const float* o2w, const float* o2b,
                      float* __restrict__ x) {
  int b = blockIdx.x / NMAP, m = blockIdx.x % NMAP;
  __shared__ float f[NPT][9], msk[NPT], h1[NPT][64], p1[64], h2[NPT][64], p2[64], t1[64];
  __shared__ float anyv;
  int t = threadIdx.x;
  for (int i = t; i < NPT * 9; i += 64)
    f[i / 9][i % 9] = feat[(size_t)(b * NMAP + m) * NPT * 9 + i];
  if (t < NPT) msk[t] = fmask[(size_t)(b * NMAP + m) * NPT + t] ? 1.f : 0.f;
  __syncthreads();
  if (t == 0) {
    float a = 0.f;
    for (int p = 0; p < NPT; ++p) a = fmaxf(a, msk[p]);
    anyv = a;
  }
  __syncthreads();
  for (int p = 0; p < NPT; ++p) {
    float acc = preb[t];
    for (int c = 0; c < 9; ++c) acc += f[p][c] * prew[c * 64 + t];
    h1[p][t] = fmaxf(acc, 0.f) * msk[p];
  }
  __syncthreads();
  float mp = -1e9f;
  for (int p = 0; p < NPT; ++p) mp = fmaxf(mp, msk[p] > 0.f ? h1[p][t] : -1e9f);
  p1[t] = anyv > 0.f ? mp : 0.f;
  __syncthreads();
  for (int p = 0; p < NPT; ++p) {
    float acc = midb[t];
    for (int c = 0; c < 64; ++c) acc += h1[p][c] * midw[c * 64 + t];
    for (int c = 0; c < 64; ++c) acc += p1[c] * midw[(64 + c) * 64 + t];
    h2[p][t] = fmaxf(acc, 0.f) * msk[p];
  }
  __syncthreads();
  float mp2 = -1e9f;
  for (int p = 0; p < NPT; ++p) mp2 = fmaxf(mp2, msk[p] > 0.f ? h2[p][t] : -1e9f);
  p2[t] = anyv > 0.f ? mp2 : 0.f;
  __syncthreads();
  float acc = o1b[t];
  for (int c = 0; c < 64; ++c) acc += p2[c] * o1w[c * 64 + t];
  t1[t] = fmaxf(acc, 0.f);
  __syncthreads();
  for (int g = 0; g < 4; ++g) {
    int d = t + g * 64;
    float a2 = o2b[d];
    for (int c = 0; c < 64; ++c) a2 += t1[c] * o2w[c * DM + d];
    x[((size_t)(b * NT + m)) * DM + d] = a2;
  }
}

// =============================================================================
// Build concatenated pos / heading / token-valid arrays (last-valid agent pose)
// =============================================================================
__global__ void build_pose_kernel(const float* mpos, const float* mhead, const u8* mvalid,
                                  const float* apos, const float* ahead, const u8* avalid,
                                  const float* lpos, const float* lhead, const u8* lvalid,
                                  float* pos, float* head, u8* tmask) {
  int i = blockIdx.x * blockDim.x + threadIdx.x;
  if (i >= TT) return;
  int b = i / NT, tk = i % NT;
  if (tk < NMAP) {
    pos[i * 2]     = mpos[(size_t)(b * NMAP + tk) * 2];
    pos[i * 2 + 1] = mpos[(size_t)(b * NMAP + tk) * 2 + 1];
    head[i]  = mhead[b * NMAP + tk];
    tmask[i] = mvalid[b * NMAP + tk] ? 1 : 0;
  } else if (tk < NMAP + NAG) {
    int n = tk - NMAP, last = 0, any = 0;
    for (int t = 0; t < THIST; ++t)
      if (avalid[(b * THIST + t) * NAG + n]) { last = t; any = 1; }
    float sc = any ? 1.f : 0.f;
    pos[i * 2]     = apos[((size_t)(b * THIST + last) * NAG + n) * 2] * sc;
    pos[i * 2 + 1] = apos[((size_t)(b * THIST + last) * NAG + n) * 2 + 1] * sc;
    head[i]  = ahead[(b * THIST + last) * NAG + n] * sc;
    tmask[i] = (u8)any;
  } else {
    int l = tk - NMAP - NAG;
    pos[i * 2]     = lpos[(size_t)(b * NLGT + l) * 2];
    pos[i * 2 + 1] = lpos[(size_t)(b * NLGT + l) * 2 + 1];
    head[i] = lhead[b * NLGT + l];
    int any = 0;
    for (int t = 0; t < THIST; ++t) any |= lvalid[(b * THIST + t) * NLGT + l] ? 1 : 0;
    tmask[i] = (u8)any;
  }
}

// =============================================================================
// Top-128 nearest valid neighbors (jax.lax.top_k semantics: value desc, tie ->
// lower index). 128 argmax rounds over LDS, one block per token.
// =============================================================================
__global__ __launch_bounds__(256)
void topk_kernel(const float* __restrict__ pos, const u8* __restrict__ tmask,
                 int* __restrict__ idx) {
  int b = blockIdx.x / NT, i = blockIdx.x % NT;
  __shared__ float negd[NT];
  __shared__ float rv[256];
  __shared__ int ri[256];
  int t = threadIdx.x;
  float px = pos[(size_t)(b * NT + i) * 2], py = pos[(size_t)(b * NT + i) * 2 + 1];
  int mi = tmask[b * NT + i];
  for (int j = t; j < NT; j += 256) {
    float dx = pos[(size_t)(b * NT + j) * 2] - px;
    float dy = pos[(size_t)(b * NT + j) * 2 + 1] - py;
    float d = sqrtf(dx * dx + dy * dy + 1e-12f);
    negd[j] = (mi && tmask[b * NT + j]) ? -d : -1e30f;
  }
  __syncthreads();
  for (int r = 0; r < KNN_; ++r) {
    float bv = -3.0e38f;
    int bj = 1 << 30;
    for (int j = t; j < NT; j += 256) {
      float vv = negd[j];
      if (vv > bv || (vv == bv && j < bj)) { bv = vv; bj = j; }
    }
    rv[t] = bv; ri[t] = bj;
    __syncthreads();
    for (int s = 128; s > 0; s >>= 1) {
      if (t < s) {
        float v2 = rv[t + s]; int j2 = ri[t + s];
        if (v2 > rv[t] || (v2 == rv[t] && j2 < ri[t])) { rv[t] = v2; ri[t] = j2; }
      }
      __syncthreads();
    }
    if (t == 0) {
      idx[(size_t)(b * NT + i) * KNN_ + r] = ri[0];
      negd[ri[0]] = -3.4e38f;
    }
    __syncthreads();
  }
}

// =============================================================================
// Relation PE: rotate rel pos into query frame, interleaved sin/cos embedding,
// masked by relm; stored f16 (reused by all 3 layers as kg/vg additive PE).
// One block per token, one thread per neighbor (writes 512B contiguous).
// =============================================================================
__global__ __launch_bounds__(KNN_)
void pe_kernel(const float* __restrict__ pos, const float* __restrict__ head,
               const u8* __restrict__ tmask, const int* __restrict__ idx,
               u8* __restrict__ relm, half_t* __restrict__ pe) {
  int bi = blockIdx.x;
  int b = bi / NT, i = bi % NT;
  int kk = threadIdx.x;
  int j = idx[(size_t)bi * KNN_ + kk];
  float px = pos[(size_t)(b * NT + i) * 2], py = pos[(size_t)(b * NT + i) * 2 + 1];
  float hx = head[b * NT + i];
  float relx = pos[(size_t)(b * NT + j) * 2] - px;
  float rely = pos[(size_t)(b * NT + j) * 2 + 1] - py;
  float dist = sqrtf(relx * relx + rely * rely + 1e-12f);
  int rm = (tmask[b * NT + i] && tmask[b * NT + j] && dist < 100.0f) ? 1 : 0;
  relm[(size_t)bi * KNN_ + kk] = (u8)rm;
  float ang = -(hx - 1.5707963267948966f);
  float c = cosf(ang), s = sinf(ang);
  float rx = c * relx - s * rely;
  float ry = s * relx + c * rely;
  float ph = head[b * NT + j] - hx;
  float sc = rm ? 1.f : 0.f;
  half_t* out = pe + ((size_t)bi * KNN_ + kk) * DM;
  const float twopi = 6.283185307179586f;
  for (int d = 0; d < 64; ++d) {
    float dim_t = powf(10000.f, (float)(d & ~1) / 64.f);
    float e = rx * twopi / dim_t;
    out[d] = (half_t)(((d & 1) ? cosf(e) : sinf(e)) * sc);
  }
  for (int d = 0; d < 64; ++d) {
    float dim_t = powf(10000.f, (float)(d & ~1) / 64.f);
    float e = ry * twopi / dim_t;
    out[64 + d] = (half_t)(((d & 1) ? cosf(e) : sinf(e)) * sc);
  }
  for (int d = 0; d < 128; ++d) {
    float dim_t = powf(10000.f, (float)(d & ~1) / 128.f);
    float e = ph * twopi / dim_t;
    out[128 + d] = (half_t)(((d & 1) ? cosf(e) : sinf(e)) * sc);
  }
}

// =============================================================================
// Gathered-neighbor attention: one 128-thread block per token.
// scores[h,k] = q_h . (k[idx]+pe)_h / sqrt(32); masked softmax; o = attn @ (v[idx]+pe)
// =============================================================================
__global__ __launch_bounds__(KNN_)
void attn_kernel(const float* __restrict__ q, const float* __restrict__ k,
                 const float* __restrict__ v, const half_t* __restrict__ pe,
                 const int* __restrict__ idx, const u8* __restrict__ relm,
                 float* __restrict__ o) {
  int bi = blockIdx.x;
  int b = bi / NT;
  __shared__ float qs[DM];
  __shared__ float sc[8][KNN_];
  __shared__ int js[KNN_];
  __shared__ float mxs[8], isum[8];
  int t = threadIdx.x;
  qs[t]       = q[(size_t)bi * DM + t];
  qs[t + 128] = q[(size_t)bi * DM + t + 128];
  int j = idx[(size_t)bi * KNN_ + t];
  js[t] = j;
  int rm = relm[(size_t)bi * KNN_ + t];
  __syncthreads();
  const float*  krow  = k + ((size_t)(b * NT) + j) * DM;
  const half_t* perow = pe + ((size_t)bi * KNN_ + t) * DM;
#pragma unroll
  for (int h = 0; h < 8; ++h) {
    float a = 0.f;
    for (int d = 0; d < 32; ++d) {
      int dd = h * 32 + d;
      a += qs[dd] * (krow[dd] + (float)perow[dd]);
    }
    sc[h][t] = rm ? a * 0.17677669529663687f : -1e9f;
  }
  __syncthreads();
  if (t < 8) {
    float mx = -3.4e38f;
    for (int n2 = 0; n2 < KNN_; ++n2) mx = fmaxf(mx, sc[t][n2]);
    mxs[t] = mx;
  }
  __syncthreads();
#pragma unroll
  for (int h = 0; h < 8; ++h) sc[h][t] = expf(sc[h][t] - mxs[h]);
  __syncthreads();
  if (t < 8) {
    float s = 0.f;
    for (int n2 = 0; n2 < KNN_; ++n2) s += sc[t][n2];
    isum[t] = 1.f / s;
  }
  __syncthreads();
  float a0 = 0.f, a1 = 0.f;
  int h0 = t >> 5, h1 = (t + 128) >> 5;
  for (int kk = 0; kk < KNN_; ++kk) {
    const float*  vrow = v + ((size_t)(b * NT) + js[kk]) * DM;
    const half_t* pr   = pe + ((size_t)bi * KNN_ + kk) * DM;
    a0 += sc[h0][kk] * (vrow[t] + (float)pr[t]);
    a1 += sc[h1][kk] * (vrow[t + 128] + (float)pr[t + 128]);
  }
  o[(size_t)bi * DM + t]       = a0 * isum[h0];
  o[(size_t)bi * DM + t + 128] = a1 * isum[h1];
}

// =============================================================================
// LayerNorm over last dim (256): one block per row
// =============================================================================
__global__ __launch_bounds__(256)
void ln_kernel(const float* __restrict__ tin, const float* __restrict__ g,
               const float* __restrict__ bta, float* __restrict__ xout) {
  int row = blockIdx.x, t = threadIdx.x;
  __shared__ float red[256];
  __shared__ float mean_s, var_s;
  float v = tin[(size_t)row * DM + t];
  red[t] = v;
  __syncthreads();
  for (int s = 128; s > 0; s >>= 1) { if (t < s) red[t] += red[t + s]; __syncthreads(); }
  if (t == 0) mean_s = red[0] * (1.f / DM);
  __syncthreads();
  float dv = v - mean_s;
  red[t] = dv * dv;
  __syncthreads();
  for (int s = 128; s > 0; s >>= 1) { if (t < s) red[t] += red[t + s]; __syncthreads(); }
  if (t == 0) var_s = red[0] * (1.f / DM);
  __syncthreads();
  xout[(size_t)row * DM + t] = dv * rsqrtf(var_s + 1e-5f) * g[t] + bta[t];
}

// =============================================================================
// Host: workspace layout + launch schedule
// =============================================================================
extern "C" void kernel_launch(void* const* d_in, const int* in_sizes, int n_in,
                              void* d_out, int out_size, void* d_ws, size_t ws_size,
                              hipStream_t stream) {
  (void)in_sizes; (void)n_in; (void)out_size; (void)ws_size;
  #define FP(i)  ((const float*)d_in[(i)])
  #define U8P(i) ((const u8*)d_in[(i)])
  #define IP(i)  ((const int*)d_in[(i)])

  // ---- workspace carve-up (bytes, 256-aligned) ----
  char* base = (char*)d_ws;
  size_t off = 0;
  auto carve = [&](size_t bytes) { char* p = base + off; off = (off + bytes + 255) & ~(size_t)255; return (void*)p; };
  const size_t LW = 786432;                       // packed halfs per layer
  half_t* wpk   = (half_t*)carve((3 * LW + 65536) * sizeof(half_t));
  float*  xb    = (float*)carve((size_t)TT * DM * 4);
  float*  qb    = (float*)carve((size_t)TT * DM * 4);
  float*  kb    = (float*)carve((size_t)TT * DM * 4);
  float*  vb    = (float*)carve((size_t)TT * DM * 4);
  float*  ob    = (float*)carve((size_t)TT * DM * 4);
  float*  tb    = (float*)carve((size_t)TT * DM * 4);
  float*  hb    = (float*)carve((size_t)TT * 1024 * 4);
  float*  posb  = (float*)carve((size_t)TT * 2 * 4);
  float*  headb = (float*)carve((size_t)TT * 4);
  u8*     tmb   = (u8*)carve(TT);
  int*    idxb  = (int*)carve((size_t)TT * KNN_ * 4);
  u8*     relmb = (u8*)carve((size_t)TT * KNN_);
  half_t* peb   = (half_t*)carve((size_t)TT * KNN_ * DM * sizeof(half_t));

  auto pack = [&](const float* src, half_t* dst, int Kd, int Nd) {
    int tot = Kd * Nd;
    pack_w_kernel<<<(tot + 255) / 256, 256, 0, stream>>>(src, dst, Kd, Nd);
  };
  // ---- pack encoder weights to WMMA fragment layout ----
  for (int l = 0; l < 3; ++l) {
    int p0 = IN_LAYER0 + 16 * l;
    pack(FP(p0 + 0),  wpk + l * LW + 0,      DM, DM);      // wq
    pack(FP(p0 + 2),  wpk + l * LW + 65536,  DM, DM);      // wk
    pack(FP(p0 + 4),  wpk + l * LW + 131072, DM, DM);      // wv
    pack(FP(p0 + 6),  wpk + l * LW + 196608, DM, DM);      // wo
    pack(FP(p0 + 10), wpk + l * LW + 262144, DM, 1024);    // w1
    pack(FP(p0 + 12), wpk + l * LW + 524288, 1024, DM);    // w2
  }
  pack(FP(IN_OUT_W), wpk + 3 * LW, DM, DM);

  // ---- token embedding (writes x at per-type offsets) ----
  map_embed_kernel<<<BB * NMAP, 64, 0, stream>>>(
      FP(IN_MAP_FEAT), U8P(IN_MAP_FVALID),
      FP(IN_MAP_PRE_W), FP(IN_MAP_PRE_B), FP(IN_MAP_MID_W), FP(IN_MAP_MID_B),
      FP(IN_MAP_O1_W), FP(IN_MAP_O1_B), FP(IN_MAP_O2_W), FP(IN_MAP_O2_B), xb);
  agent_embed_kernel<<<BB * NAG, 256, 0, stream>>>(
      FP(IN_AGENT_FEAT), U8P(IN_AGENT_VALID), IP(IN_AGENT_ID),
      FP(IN_AG_W0), FP(IN_AG_B0), FP(IN_AG_W1), FP(IN_AG_B1),
      FP(IN_AG_W2), FP(IN_AG_B2), FP(IN_AGENT_PE), xb);
  light_embed_kernel<<<BB * NLGT, 256, 0, stream>>>(
      FP(IN_TL_FEAT), U8P(IN_TL_VALID),
      FP(IN_LT_W0), FP(IN_LT_B0), FP(IN_LT_W1), FP(IN_LT_B1),
      FP(IN_LT_W2), FP(IN_LT_B2), xb);

  // ---- relation graph ----
  build_pose_kernel<<<(TT + 255) / 256, 256, 0, stream>>>(
      FP(IN_MAP_POS), FP(IN_MAP_HEAD), U8P(IN_MAP_VALID),
      FP(IN_AGENT_POS), FP(IN_AGENT_HEAD), U8P(IN_AGENT_VALID),
      FP(IN_TL_POS), FP(IN_TL_HEAD), U8P(IN_TL_VALID),
      posb, headb, tmb);
  topk_kernel<<<TT, 256, 0, stream>>>(posb, tmb, idxb);
  pe_kernel<<<TT, KNN_, 0, stream>>>(posb, headb, tmb, idxb, relmb, peb);

  auto gemm = [&](const float* A, const half_t* Wp, const float* bias,
                  const float* res, const u8* rmask, float* C,
                  int Nd, int Kd, int flags) {
    dim3 g(TT / 16, Nd / 64);
    gemm_wmma_kernel<<<g, 32, 0, stream>>>(A, Wp, bias, res, rmask, C, Nd, Kd, flags);
  };

  // ---- encoder layers ----
  for (int l = 0; l < 3; ++l) {
    int p0 = IN_LAYER0 + 16 * l;
    half_t* pk = wpk + l * LW;
    gemm(xb, pk + 0,      FP(p0 + 1), nullptr, nullptr, qb, DM, DM, 0);
    gemm(xb, pk + 65536,  FP(p0 + 3), nullptr, nullptr, kb, DM, DM, 0);
    gemm(xb, pk + 131072, FP(p0 + 5), nullptr, nullptr, vb, DM, DM, 0);
    attn_kernel<<<TT, KNN_, 0, stream>>>(qb, kb, vb, peb, idxb, relmb, ob);
    gemm(ob, pk + 196608, FP(p0 + 7), xb, nullptr, tb, DM, DM, 2);          // +residual
    ln_kernel<<<TT, 256, 0, stream>>>(tb, FP(p0 + 8), FP(p0 + 9), xb);
    gemm(xb, pk + 262144, FP(p0 + 11), nullptr, nullptr, hb, 1024, DM, 1);  // ReLU
    gemm(hb, pk + 524288, FP(p0 + 13), xb, nullptr, tb, DM, 1024, 2);       // +residual
    ln_kernel<<<TT, 256, 0, stream>>>(tb, FP(p0 + 14), FP(p0 + 15), xb);
  }

  // ---- output projection * token mask ----
  gemm(xb, wpk + 3 * LW, FP(IN_OUT_B), nullptr, tmb, (float*)d_out, DM, DM, 4);
  #undef FP
  #undef U8P
  #undef IP
}
